// tNN_29068338659450
// MI455X (gfx1250) — matrix operand order repaired and checked by reading, not compiled, not run on hardware
//
#include <hip/hip_runtime.h>
#include <hip/hip_bf16.h>
#include <stdint.h>

// ---------------- types ----------------
typedef __bf16        v16bf __attribute__((ext_vector_type(16)));
typedef float         v8f   __attribute__((ext_vector_type(8)));
typedef unsigned int  u32x4 __attribute__((ext_vector_type(4)));
typedef int           i32x8 __attribute__((ext_vector_type(8)));
typedef int           i32x4 __attribute__((ext_vector_type(4)));

#define LTOT      28
#define BATCH_P   32768
#define WG_COLS   128          // batch columns per workgroup
#define ROWSTRIDE 129          // dwords per LDS tile row (128 + 1 pad dword, via TDM pad)
#define TILE_DW   (32 * ROWSTRIDE)   // 32 rows (28 real + 4 zero) = 4128 dwords
#define TILE_BYTES (TILE_DW * 4)     // 16512 B
#define SMEM_BYTES (2 * TILE_BYTES + 128)

// LDS byte offset of a generic pointer that points into shared memory
__device__ __forceinline__ unsigned lds_off_of(void* p) {
  return (unsigned)(uintptr_t)(__attribute__((address_space(3))) char*)p;
}

// ---- Tensor Data Mover: 2D tile load (28 rows x 128 cols of fp32) -> LDS ----
// D# per CDNA5 ISA. pad_enable inserts 1 dword after every
// 128 dwords -> LDS row stride of 129 dwords (bank-conflict-free B reads).
__device__ __forceinline__ void tdm_load_tile(unsigned lds_byte_off, const float* gsrc) {
  unsigned long long ga = (unsigned long long)(uintptr_t)gsrc;
  u32x4 g0;
  g0[0] = 1u;                                            // count=1, user mode
  g0[1] = lds_byte_off;                                  // lds_addr
  g0[2] = (unsigned)ga;                                  // global_addr[31:0]
  g0[3] = (unsigned)((ga >> 32) & 0x01FFFFFFu) | (2u << 30); // addr[56:32], type=2

  i32x8 g1;
  g1[0] = (2 << 16)      // data_size = 4 bytes
        | (1 << 20)      // pad_enable
        | (6 << 22);     // pad_interval = 128 dwords; pad_amount field = 0 -> 1 dword
  g1[1] = (int)((BATCH_P & 0xFFFF) << 16);                       // tensor_dim0 lo16
  g1[2] = (int)(((BATCH_P >> 16) & 0xFFFF) | (LTOT << 16));      // td0 hi16 | tensor_dim1 lo16
  g1[3] = (int)(WG_COLS << 16);                                  // td1 hi16=0 | tile_dim0=128
  g1[4] = (int)(LTOT);                                           // tile_dim1=28, tile_dim2=0
  g1[5] = (int)(BATCH_P);                                        // tensor_dim0_stride lo32
  g1[6] = 0;                                                     // stride0 hi | stride1 lo
  g1[7] = 0;

  i32x4 z4;
  z4[0] = 0; z4[1] = 0; z4[2] = 0; z4[3] = 0;
  i32x8 z8;
#pragma unroll
  for (int e = 0; e < 8; ++e) z8[e] = 0;
  // 6-arg variant (clang-23 / therock-10.0 headers)
  __builtin_amdgcn_tensor_load_to_lds(g0, g1, z4, z4, z8, 0);
}

// ---- prep: swizzle W into per-lane bf16 A-operand layout ----
// dst layout: [(layer*28 + j')][mt][lane][i], 16 bf16 (32B) per lane, K padded to 32,
// rows m >= M_out zero-padded.  A 16-bit A-matrix element map (ISA 7.12.2):
//   lane 0-15: M = lane,  i=0..7 -> K=i, i=8..15 -> K=i+8
//   lane16-31: M = lane-16, same but K += 8
__global__ __launch_bounds__(64) void prep_weights(
    const float* __restrict__ W1, const float* __restrict__ W2,
    const float* __restrict__ W3, const float* __restrict__ W4,
    __bf16* __restrict__ wout) {
  int b  = blockIdx.x;            // 0..111  (layer*28 + j')
  int sl = b / 28, jp = b % 28;
  const float* W = (sl == 0) ? W1 : (sl == 1) ? W2 : (sl == 2) ? W3 : W4;
  int Mo = (sl == 3) ? 10 : 28;
  __bf16* dst = wout + (size_t)b * (2 * 32 * 16);
  for (int idx = threadIdx.x; idx < 2 * 32 * 16; idx += 64) {
    int mt   = idx >> 9;
    int lane = (idx >> 4) & 31;
    int i    = idx & 15;
    int m    = mt * 16 + (lane & 15);
    int kk   = i + ((i >= 8) ? 8 : 0) + ((lane >= 16) ? 8 : 0);
    float v  = 0.0f;
    if (m < Mo && kk < 28) v = W[(size_t)jp * Mo * 28 + m * 28 + kk];
    dst[(mt * 32 + lane) * 16 + i] = (__bf16)v;
  }
}

// ---- layer kernel: out[k,:,cols] = relu( sum_j W[(k-j)%28] @ in[j,:,cols] + bias[k] )
template <int M_OUT, int MT>
__global__ __launch_bounds__(128) void tprod_layer(
    const float* __restrict__ in,     // (28, 28, P) fp32
    const __bf16* __restrict__ wswz,  // swizzled weights for this layer
    const float* __restrict__ bias,   // (28, M_OUT)
    float* __restrict__ out) {        // (28, M_OUT, P) fp32
  extern __shared__ char smem[];
  float* tile0   = (float*)smem;
  float* tile1   = (float*)(smem + TILE_BYTES);
  float* biasLds = (float*)(smem + 2 * TILE_BYTES);

  const int k      = blockIdx.y;
  const int col0   = blockIdx.x * WG_COLS;
  const int tid    = threadIdx.x;
  const int lane   = tid & 31;
  const int wave   = tid >> 5;
  const int lanelo = lane & 15;
  const int hi     = (lane >> 4) & 1;

  // zero the K-pad rows (28..31) of both tiles; stage bias
  for (int i = tid; i < 4 * ROWSTRIDE; i += 128) {
    tile0[28 * ROWSTRIDE + i] = 0.0f;
    tile1[28 * ROWSTRIDE + i] = 0.0f;
  }
  if (tid < 32) biasLds[tid] = (tid < M_OUT) ? bias[k * M_OUT + tid] : 0.0f;
  __syncthreads();

  const unsigned t0off = lds_off_of(tile0);
  const unsigned t1off = lds_off_of(tile1);

  if (wave == 0) {  // wave 0 drives the TDM double-buffer pipeline
    tdm_load_tile(t0off, in + (size_t)0 * 28 * BATCH_P + col0);
    tdm_load_tile(t1off, in + (size_t)1 * 28 * BATCH_P + col0);
  }

  v8f zero8;
#pragma unroll
  for (int e = 0; e < 8; ++e) zero8[e] = 0.0f;
  v8f acc[MT][2];
#pragma unroll
  for (int mt = 0; mt < MT; ++mt)
#pragma unroll
    for (int nt = 0; nt < 2; ++nt) acc[mt][nt] = zero8;

  const int khalf = hi ? 8 : 0;            // B operand: lanes 16-31 hold K=8..15,24..31
  const int ncol  = wave * 32 + lanelo;    // this wave's 32-column slice

  for (int j = 0; j < 28; ++j) {
    if (wave == 0) {
      if (j < 27) __builtin_amdgcn_s_wait_tensorcnt(1);   // tile j landed
      else        __builtin_amdgcn_s_wait_tensorcnt(0);
    }
    __syncthreads();
    const float* lbuf = (j & 1) ? tile1 : tile0;

    int jp = k - j; if (jp < 0) jp += 28;                 // circulant index
    const __bf16* wj = wswz + (size_t)jp * (2 * 32 * 16);

    v16bf a[MT];
#pragma unroll
    for (int mt = 0; mt < MT; ++mt)
      a[mt] = *(const v16bf*)(wj + (mt * 32 + lane) * 16);

#pragma unroll
    for (int nt = 0; nt < 2; ++nt) {
      v16bf bmat;
#pragma unroll
      for (int i = 0; i < 16; ++i)
        bmat[i] = (__bf16)lbuf[(khalf + i + ((i >= 8) ? 8 : 0)) * ROWSTRIDE + ncol + nt * 16];
#pragma unroll
      for (int mt = 0; mt < MT; ++mt)
        acc[mt][nt] = __builtin_amdgcn_wmma_f32_16x16x32_bf16(
            false, a[mt], false, bmat, (short)0, acc[mt][nt], false, false);
    }

    __syncthreads();  // everyone done reading buf[j&1]
    if (wave == 0 && j + 2 < 28)
      tdm_load_tile((j & 1) ? t1off : t0off, in + (size_t)(j + 2) * 28 * BATCH_P + col0);
  }

  // epilogue: bias + relu + store (C/D layout: lanes0-15 M=vg, lanes16-31 M=vg+8)
#pragma unroll
  for (int mt = 0; mt < MT; ++mt)
#pragma unroll
    for (int nt = 0; nt < 2; ++nt)
#pragma unroll
      for (int vg = 0; vg < 8; ++vg) {
        int m = mt * 16 + vg + hi * 8;
        float v = acc[mt][nt][vg] + biasLds[m];
        v = (v > 0.0f) ? v : 0.0f;
        if (m < M_OUT) {
          int col = col0 + wave * 32 + nt * 16 + lanelo;
          out[(size_t)(k * M_OUT + m) * BATCH_P + col] = v;
        }
      }
}

// ---------------- host launcher ----------------
extern "C" void kernel_launch(void* const* d_in, const int* in_sizes, int n_in,
                              void* d_out, int out_size, void* d_ws, size_t ws_size,
                              hipStream_t stream) {
  (void)in_sizes; (void)n_in; (void)out_size; (void)ws_size;
  const float* x  = (const float*)d_in[0];
  const float* W1 = (const float*)d_in[1];
  const float* B1 = (const float*)d_in[2];
  const float* W2 = (const float*)d_in[3];
  const float* B2 = (const float*)d_in[4];
  const float* W3 = (const float*)d_in[5];
  const float* B3 = (const float*)d_in[6];
  const float* W4 = (const float*)d_in[7];
  const float* B4 = (const float*)d_in[8];
  float* out = (float*)d_out;

  char* ws = (char*)d_ws;
  __bf16* wswz = (__bf16*)ws;                       // 4*28*2*32*16 bf16 = 229376 B
  size_t wbytes = (size_t)4 * 28 * 2 * 32 * 16 * sizeof(__bf16);
  float* act0 = (float*)(ws + wbytes);              // (28,28,P) fp32 = 102.8 MB
  float* act1 = act0 + (size_t)28 * 28 * BATCH_P;   // ping-pong

  prep_weights<<<112, 64, 0, stream>>>(W1, W2, W3, W4, wswz);

  dim3 grid(BATCH_P / WG_COLS, LTOT);               // (256, 28)
  const size_t sm = SMEM_BYTES;
  const size_t wlstride = (size_t)28 * 2 * 32 * 16; // bf16 elems per layer

  tprod_layer<28, 2><<<grid, 128, sm, stream>>>(x,    wswz + 0 * wlstride, B1, act0);
  tprod_layer<28, 2><<<grid, 128, sm, stream>>>(act0, wswz + 1 * wlstride, B2, act1);
  tprod_layer<28, 2><<<grid, 128, sm, stream>>>(act1, wswz + 2 * wlstride, B3, act0);
  tprod_layer<10, 1><<<grid, 128, sm, stream>>>(act0, wswz + 3 * wlstride, B4, out);
}
